// CheckinScorer_52862457479737
// MI455X (gfx1250) — matrix operand order repaired and checked by reading, not compiled, not run on hardware
//
#include <hip/hip_runtime.h>
#include <hip/hip_bf16.h>

#define HH 128

typedef __attribute__((ext_vector_type(16))) __bf16 v16bf;
typedef __attribute__((ext_vector_type(8)))  float  v8f;

// ---------------- helpers: ordered-uint encoding for float atomicMax --------
__device__ __forceinline__ unsigned enc_f(float f) {
    unsigned b = __float_as_uint(f);
    return (b & 0x80000000u) ? ~b : (b | 0x80000000u);
}
__device__ __forceinline__ float dec_f(unsigned e) {
    if (e == 0u) return 0.0f;                 // sentinel: empty segment -> m=0
    unsigned b = (e & 0x80000000u) ? (e & 0x7FFFFFFFu) : ~e;
    return __uint_as_float(b);
}

// ---------------- weight transpose + f32->bf16 convert ----------------------
// W is [K,128] row-major f32; Wt is [128,Kp] row-major bf16 (zero padded)
__global__ void conv_wt_kernel(const float* __restrict__ W, __bf16* __restrict__ Wt,
                               int K, int Kp) {
    int idx = blockIdx.x * blockDim.x + threadIdx.x;
    int total = HH * Kp;
    if (idx >= total) return;
    int n = idx / Kp, k = idx - n * Kp;
    Wt[idx] = (k < K) ? (__bf16)W[(size_t)k * HH + n] : (__bf16)0.0f;
}

// ---------------- a_edge coefficient: dot(lin_edge_w, att_edge) -------------
__global__ void dot128_kernel(const float* __restrict__ a, const float* __restrict__ b,
                              float* __restrict__ out) {
    int lane = threadIdx.x & 31;
    float s = 0.f;
#pragma unroll
    for (int j = 0; j < 4; ++j) { int k = lane + 32 * j; s += a[k] * b[k]; }
#pragma unroll
    for (int off = 16; off; off >>= 1) s += __shfl_xor(s, off, 32);
    if (lane == 0) out[0] = s;
}

// ---------------- row gather: dst[i,:] = src[idx[i],:] ----------------------
__global__ void gather_rows_kernel(const float* __restrict__ src, const int* __restrict__ idx,
                                   float* __restrict__ dst, int N) {
    int t = blockIdx.x * blockDim.x + threadIdx.x;
    if (t >= N * HH) return;
    int r = t >> 7, c = t & 127;
    dst[t] = src[(size_t)idx[r] * HH + c];
}

// ---------------- WMMA GEMM: C[N,128] = X[N,K] @ W[K,128] (+bias +addmat) ---
// Wt: bf16 [128,Kp] (n-major); A fragments hold contiguous k (K=j+16*(lane>=16)), same order as B,
// B fragments are contiguous 32B loads per lane (N=lane%16, K=j+16*(lane>=16)).
template <int KC, bool EXACTK>
__global__ __launch_bounds__(256) void gemm_xw_kernel(
    const float* __restrict__ X, const __bf16* __restrict__ Wt,
    float* __restrict__ C, int N, int K,
    const float* __restrict__ bias, const float* __restrict__ addmat,
    const int* __restrict__ addidx) {
    const int lane = threadIdx.x & 31;
    const int wave = threadIdx.x >> 5;
    const int m0 = blockIdx.x * 128 + wave * 16;

    // ---- load A fragments (all K chunks) ----
    const int arow = m0 + (lane & 15);
    const bool arok = arow < N;
    const float* xr = X + (size_t)(arok ? arow : 0) * K;
    const int kb0 = (lane >> 4) << 4;   // 0 or 16

    v16bf A[KC];
#pragma unroll
    for (int c = 0; c < KC; ++c) {
        float t[16];
        const int kb = c * 32 + kb0;
        if (EXACTK) {
            const float4 p0 = *(const float4*)(xr + kb);
            const float4 p1 = *(const float4*)(xr + kb + 4);
            const float4 p2 = *(const float4*)(xr + kb + 8);
            const float4 p3 = *(const float4*)(xr + kb + 12);
            t[0]=p0.x; t[1]=p0.y; t[2]=p0.z; t[3]=p0.w;
            t[4]=p1.x; t[5]=p1.y; t[6]=p1.z; t[7]=p1.w;
            t[8]=p2.x; t[9]=p2.y; t[10]=p2.z; t[11]=p2.w;
            t[12]=p3.x; t[13]=p3.y; t[14]=p3.z; t[15]=p3.w;
        } else {
#pragma unroll
            for (int j = 0; j < 8; ++j) {
                int k1 = kb + j, k2 = kb + 8 + j;
                t[j]     = (k1 < K) ? xr[k1] : 0.f;
                t[8 + j] = (k2 < K) ? xr[k2] : 0.f;
            }
        }
#pragma unroll
        for (int j = 0; j < 16; ++j) A[c][j] = arok ? (__bf16)t[j] : (__bf16)0.f;
    }

    // ---- 8 column tiles of 16, K accumulated via v_wmma_f32_16x16x32_bf16 ----
    const int nlo = lane & 15;
    const int bko = (lane >> 4) << 4;   // 0 or 16
    const int Kp = KC * 32;
    const int crow0 = m0 + 8 * (lane >> 4);
#pragma unroll
    for (int t = 0; t < 8; ++t) {
        v8f acc = {0.f, 0.f, 0.f, 0.f, 0.f, 0.f, 0.f, 0.f};
        const int n = t * 16 + nlo;
        const __bf16* wrow = Wt + (size_t)n * Kp + bko;
#pragma unroll
        for (int c = 0; c < KC; ++c) {
            v16bf B = *(const v16bf*)(wrow + c * 32);
            acc = __builtin_amdgcn_wmma_f32_16x16x32_bf16(
                false, A[c], false, B, (short)0, acc, false, false);
        }
        const float badd = bias ? bias[n] : 0.f;
#pragma unroll
        for (int v = 0; v < 8; ++v) {
            int r = crow0 + v;
            if (r < N) {
                float val = acc[v] + badd;
                if (addmat) {
                    int ai = addidx ? addidx[r] : r;
                    val += addmat[(size_t)ai * HH + n];
                }
                C[(size_t)r * HH + n] = val;
            }
        }
    }
}

// ---------------- per-row dot with attention vector -------------------------
__global__ __launch_bounds__(256) void rowdot_kernel(const float* __restrict__ Hm,
                                                     const float* __restrict__ att,
                                                     float* __restrict__ out, int N) {
    int lane = threadIdx.x & 31, wv = threadIdx.x >> 5;
    int row = blockIdx.x * 8 + wv;
    if (row >= N) return;
    const float* hr = Hm + (size_t)row * HH;
    float s = 0.f;
#pragma unroll
    for (int j = 0; j < 4; ++j) { int k = lane + 32 * j; s += hr[k] * att[k]; }
#pragma unroll
    for (int off = 16; off; off >>= 1) s += __shfl_xor(s, off, 32);
    if (lane == 0) out[row] = s;
}

// ---------------- edge kernels ----------------------------------------------
__global__ void edge_logits_kernel(const int* __restrict__ src, const int* __restrict__ dst,
                                   const float* __restrict__ eattr,
                                   const float* __restrict__ aecoef,
                                   const float* __restrict__ asrc, const float* __restrict__ adst,
                                   float* __restrict__ elog, unsigned* __restrict__ menc, int E) {
    int e = blockIdx.x * blockDim.x + threadIdx.x;
    if (e >= E) return;
    float l = asrc[src[e]] + adst[dst[e]] + aecoef[0] * eattr[e];
    l = (l >= 0.f) ? l : 0.2f * l;     // leaky_relu 0.2
    elog[e] = l;
    atomicMax(&menc[dst[e]], enc_f(l));
}

__global__ void edge_exp_kernel(const int* __restrict__ dst, float* __restrict__ elog,
                                const unsigned* __restrict__ menc, float* __restrict__ den, int E) {
    int e = blockIdx.x * blockDim.x + threadIdx.x;
    if (e >= E) return;
    int d = dst[e];
    float ex = expf(elog[e] - dec_f(menc[d]));
    elog[e] = ex;                       // reuse buffer in place
    atomicAdd(&den[d], ex);
}

__global__ __launch_bounds__(256) void edge_scatter_kernel(
    const int* __restrict__ src, const int* __restrict__ dst,
    const float* __restrict__ ex, const float* __restrict__ den,
    const float* __restrict__ Hm, float* __restrict__ out, int E) {
    int lane = threadIdx.x & 31, wv = threadIdx.x >> 5;
    int e = blockIdx.x * 8 + wv;
    if (e >= E) return;
    int s = src[e], d = dst[e];
    float alpha = ex[e] / (den[d] + 1e-16f);
    const float4 h4 = ((const float4*)(Hm + (size_t)s * HH))[lane];
    float* o = out + (size_t)d * HH + lane * 4;
    atomicAdd(o + 0, h4.x * alpha);
    atomicAdd(o + 1, h4.y * alpha);
    atomicAdd(o + 2, h4.z * alpha);
    atomicAdd(o + 3, h4.w * alpha);
}

__global__ void bias_act_kernel(float* __restrict__ X, const float* __restrict__ b,
                                int N, int relu) {
    int t = blockIdx.x * blockDim.x + threadIdx.x;
    if (t >= N * HH) return;
    float v = X[t] + b[t & 127];
    X[t] = relu ? fmaxf(v, 0.f) : v;
}

// ---------------- cosine similarity head ------------------------------------
__global__ __launch_bounds__(256) void cosine_kernel(const float* __restrict__ U,
                                                     const float* __restrict__ V,
                                                     const int* __restrict__ eli,
                                                     float* __restrict__ out, int EL) {
    int lane = threadIdx.x & 31, wv = threadIdx.x >> 5;
    int e = blockIdx.x * 8 + wv;
    if (e >= EL) return;
    const float* u = U + (size_t)eli[e] * HH;
    const float* v = V + (size_t)eli[EL + e] * HH;
    float d = 0.f, a = 0.f, b = 0.f;
#pragma unroll
    for (int j = 0; j < 4; ++j) {
        int k = lane + 32 * j;
        float uu = u[k], vv = v[k];
        d += uu * vv; a += uu * uu; b += vv * vv;
    }
#pragma unroll
    for (int off = 16; off; off >>= 1) {
        d += __shfl_xor(d, off, 32);
        a += __shfl_xor(a, off, 32);
        b += __shfl_xor(b, off, 32);
    }
    if (lane == 0)
        out[e] = d / (fmaxf(sqrtf(a), 1e-8f) * fmaxf(sqrtf(b), 1e-8f));
}

// ---------------- host driver ------------------------------------------------
extern "C" void kernel_launch(void* const* d_in, const int* in_sizes, int n_in,
                              void* d_out, int out_size, void* d_ws, size_t ws_size,
                              hipStream_t stream) {
    const int NU = 50000, NV = 50000, E = 800000, EL = 100000, VF = 385;
    const int KPV = 416;  // 385 padded to mult of 32

    // inputs, in setup_inputs() insertion order (params flattened depth-first)
    const int*   user_node_id = (const int*)d_in[0];
    const float* venue_x      = (const float*)d_in[1];
    const int*   venue_node_id= (const int*)d_in[2];
    const int*   edge_index   = (const int*)d_in[3];
    const float* edge_attr    = (const float*)d_in[4];
    const int*   eli          = (const int*)d_in[5];
    const float* user_emb     = (const float*)d_in[6];
    const float* venue_emb    = (const float*)d_in[7];
    const float* venue_lin_w  = (const float*)d_in[8];
    const float* venue_lin_b  = (const float*)d_in[9];
    struct ConvP { const float *W, *b, *att_src, *att_dst, *lin_edge_w, *att_edge; };
    auto getconv = [&](int base) {
        ConvP p;
        p.W = (const float*)d_in[base + 0]; p.b = (const float*)d_in[base + 1];
        p.att_src = (const float*)d_in[base + 2]; p.att_dst = (const float*)d_in[base + 3];
        p.lin_edge_w = (const float*)d_in[base + 4]; p.att_edge = (const float*)d_in[base + 5];
        return p;
    };
    ConvP cp[4] = { getconv(10), getconv(16), getconv(22), getconv(28) };

    // workspace bump allocator (256B aligned)
    char* wsp = (char*)d_ws;
    auto alloc = [&](size_t bytes) -> void* {
        void* p = wsp; wsp += (bytes + 255) & ~(size_t)255; return p;
    };
    float* xu = (float*)alloc((size_t)NU * HH * 4);   // later reused as u2
    float* xv = (float*)alloc((size_t)NV * HH * 4);   // later reused as v2
    float* hA = (float*)alloc((size_t)50000 * HH * 4);
    float* hB = (float*)alloc((size_t)50000 * HH * 4);
    float* v1 = (float*)alloc((size_t)NV * HH * 4);
    float* u1 = (float*)alloc((size_t)NU * HH * 4);
    float* asrc = (float*)alloc(50000 * 4);
    float* adst = (float*)alloc(50000 * 4);
    float* elog = (float*)alloc((size_t)E * 4);       // logits then ex, in place
    unsigned* menc = (unsigned*)alloc(50000 * 4);
    float* den = (float*)alloc(50000 * 4);
    __bf16* wt_ven = (__bf16*)alloc((size_t)HH * KPV * 2);
    __bf16* wt_c[4];
    for (int i = 0; i < 4; ++i) wt_c[i] = (__bf16*)alloc((size_t)HH * HH * 2);
    float* ae = (float*)alloc(4 * 4);

    // ---- prep: weight transpose/convert, edge-attention scalars ----
    conv_wt_kernel<<<(HH * KPV + 255) / 256, 256, 0, stream>>>(venue_lin_w, wt_ven, VF, KPV);
    for (int i = 0; i < 4; ++i) {
        conv_wt_kernel<<<(HH * HH + 255) / 256, 256, 0, stream>>>(cp[i].W, wt_c[i], HH, HH);
        dot128_kernel<<<1, 32, 0, stream>>>(cp[i].lin_edge_w, cp[i].att_edge, ae + i);
    }

    // ---- node features ----
    gather_rows_kernel<<<(NU * HH + 255) / 256, 256, 0, stream>>>(user_emb, user_node_id, xu, NU);
    gemm_xw_kernel<13, false><<<(NV + 127) / 128, 256, 0, stream>>>(
        venue_x, wt_ven, xv, NV, VF, venue_lin_b, venue_emb, venue_node_id);

    const int* su = edge_index;       // checkin: user -> venue
    const int* dv = edge_index + E;

    auto runconv = [&](int ci, const float* Xs, int ns, const float* Xd, int nd,
                       const int* s, const int* d, float* outb, int relu) {
        gemm_xw_kernel<4, true><<<(ns + 127) / 128, 256, 0, stream>>>(
            Xs, wt_c[ci], hA, ns, HH, nullptr, nullptr, nullptr);
        gemm_xw_kernel<4, true><<<(nd + 127) / 128, 256, 0, stream>>>(
            Xd, wt_c[ci], hB, nd, HH, nullptr, nullptr, nullptr);
        rowdot_kernel<<<(ns + 7) / 8, 256, 0, stream>>>(hA, cp[ci].att_src, asrc, ns);
        rowdot_kernel<<<(nd + 7) / 8, 256, 0, stream>>>(hB, cp[ci].att_dst, adst, nd);
        hipMemsetAsync(menc, 0, (size_t)nd * 4, stream);
        hipMemsetAsync(den, 0, (size_t)nd * 4, stream);
        hipMemsetAsync(outb, 0, (size_t)nd * HH * 4, stream);
        edge_logits_kernel<<<(E + 255) / 256, 256, 0, stream>>>(
            s, d, edge_attr, ae + ci, asrc, adst, elog, menc, E);
        edge_exp_kernel<<<(E + 255) / 256, 256, 0, stream>>>(d, elog, menc, den, E);
        edge_scatter_kernel<<<(E + 7) / 8, 256, 0, stream>>>(s, d, elog, den, hA, outb, E);
        bias_act_kernel<<<(nd * HH + 255) / 256, 256, 0, stream>>>(outb, cp[ci].b, nd, relu);
    };

    // layer 1 (relu)
    runconv(0, xu, NU, xv, NV, su, dv, v1, 1);   // c1_checkin: user->venue
    runconv(1, xv, NV, xu, NU, dv, su, u1, 1);   // c1_rev:     venue->user
    // layer 2 (no relu); xu/xv are now free -> reuse as u2/v2
    float* v2 = xv; float* u2 = xu;
    runconv(2, u1, NU, v1, NV, su, dv, v2, 0);   // c2_checkin
    runconv(3, v1, NV, u1, NU, dv, su, u2, 0);   // c2_rev

    // link predictor: cosine similarity over label edges
    cosine_kernel<<<(EL + 7) / 8, 256, 0, stream>>>(u2, v2, eli, (float*)d_out, EL);
}
